// EBSDDI_39968965656983
// MI455X (gfx1250) — matrix operand order repaired and checked by reading, not compiled, not run on hardware
//
#include <hip/hip_runtime.h>
#include <stdint.h>

// ---------------------------------------------------------------------------
// EBSD dictionary indexing on MI455X (gfx1250, wave32, WMMA).
//   sim = qn @ dn^T  (bf16 inputs, f32 accumulate via v_wmma_f32_16x16x32_bf16)
//   fused per-tile top-10 -> candidate merge.
// Round 6: staging via GLOBAL_LOAD_ASYNC_TO_LDS_B128 (ASYNCcnt-tracked DMA,
// inline asm). No staging VGPRs -> no scratch spills; global->LDS copy for
// slab kt+1 fully overlaps the WMMA block of slab kt. One s_wait_asynccnt 0
// + one barrier per K-step.
// ---------------------------------------------------------------------------

typedef __attribute__((ext_vector_type(16))) __bf16 v16bf;
typedef __attribute__((ext_vector_type(8)))  float  v8f;

#define QN     2048
#define PN     30000
#define PIX    3600
#define KPAD   3616              // next multiple of 32 above 3600
#define KSTEPS (KPAD / 32)       // 113
#define TOPK   10
#define BM     64                // queries per block
#define BN     256               // patterns per block
#define NCHUNK ((PN + BN - 1) / BN)   // 118
#define EPSF   1e-12f
#define NEG_INF (-3.402823e38f)

// LDS layout (bytes): A0 @0 (4K), A1 @4K, B0 @8K (16K), B1 @24K; total 40K.
// After the K-loop the whole 64K region is reused as the [64][256] f32 sim tile.
#define LDS_A0 0
#define LDS_A1 4096
#define LDS_B0 8192
#define LDS_B1 24576
#define SMEM_BYTES (BM * BN * 4)   // 65536

union Frag { v16bf v; uint4 u[2]; };

// Async DMA: 16 bytes global -> LDS per lane (GLOBAL_LOAD_ASYNC_TO_LDS_B128).
// ldsByteAddr: LDS byte address (addr[31:0] of the flat shared pointer).
__device__ __forceinline__ void async_b128(unsigned ldsByteAddr, const void* gptr) {
    asm volatile("global_load_async_to_lds_b128 %0, %1, off"
                 :: "v"(ldsByteAddr), "v"(gptr)
                 : "memory");
}
__device__ __forceinline__ void wait_async0() {
    asm volatile("s_wait_asynccnt 0x0" ::: "memory");
}

__device__ __forceinline__ unsigned short f2bf(float f) {
    // round-to-nearest-even float -> bf16 bits (inputs are finite randn data)
    unsigned int x = __float_as_uint(f);
    unsigned int r = x + 0x7FFFu + ((x >> 16) & 1u);
    return (unsigned short)(r >> 16);
}

// -------------------- preprocessing: queries -------------------------------
__global__ void prep_q(const float* __restrict__ src, unsigned short* __restrict__ dst) {
    __shared__ float red[256];
    const int q = blockIdx.x;
    const int t = threadIdx.x;
    const float* row = src + (size_t)q * PIX;

    float s = 0.f;
    for (int p = t; p < PIX; p += 256) s += row[p];
    red[t] = s; __syncthreads();
    for (int o = 128; o > 0; o >>= 1) { if (t < o) red[t] += red[t + o]; __syncthreads(); }
    const float mean = red[0] * (1.0f / PIX);
    __syncthreads();

    float ss = 0.f;
    for (int p = t; p < PIX; p += 256) { float v = row[p] - mean; ss += v * v; }
    red[t] = ss; __syncthreads();
    for (int o = 128; o > 0; o >>= 1) { if (t < o) red[t] += red[t + o]; __syncthreads(); }
    const float inv = 1.0f / (sqrtf(red[0]) + EPSF);

    unsigned short* out = dst + (size_t)q * KPAD;
    for (int p = t; p < KPAD; p += 256) {
        float v = (p < PIX) ? (row[p] - mean) * inv : 0.f;
        out[p] = f2bf(v);
    }
}

// -------------------- preprocessing: dictionary ----------------------------
__global__ void prep_d(const float* __restrict__ src, unsigned short* __restrict__ dst) {
    __shared__ float red[256];
    const int n = blockIdx.x;
    const int t = threadIdx.x;
    const float* row = src + (size_t)n * PIX;

    float ss = 0.f;
    for (int p = t; p < PIX; p += 256) { float v = row[p]; ss += v * v; }
    red[t] = ss; __syncthreads();
    for (int o = 128; o > 0; o >>= 1) { if (t < o) red[t] += red[t + o]; __syncthreads(); }
    const float inv = 1.0f / (sqrtf(red[0]) + EPSF);

    unsigned short* out = dst + (size_t)n * KPAD;
    for (int p = t; p < KPAD; p += 256) {
        float v = (p < PIX) ? row[p] * inv : 0.f;
        out[p] = f2bf(v);
    }
}

// -------------------- WMMA GEMM + fused per-tile top-k ---------------------
// Block: 256 threads = 8 waves. Output tile: 64 queries x 256 patterns.
// Wave w: rows [(w&1)*32, +32), cols [(w>>1)*64, +64): 2 A-frags x 4 B-frags.
__global__ __launch_bounds__(256, 1) void gemm_topk(
        const unsigned short* __restrict__ qbf,
        const unsigned short* __restrict__ dbf,
        float* __restrict__ candV, int* __restrict__ candI) {
    __shared__ __align__(16) unsigned char smem[SMEM_BYTES];   // 64 KB
    float* simLds = (float*)smem;                              // [64][256] (reused)

    const int t    = threadIdx.x;
    const int lane = t & 31;
    const int wave = t >> 5;
    const int qBase = blockIdx.y * BM;
    const int nBase = blockIdx.x * BN;

    const int mq = wave & 1;   // row half: m-tiles {2mq, 2mq+1}
    const int ng = wave >> 1;  // 64-column group (0..3)

    // LDS byte address of smem (flat shared addr low 32 bits == LDS offset)
    const unsigned smemAddr = (unsigned)(unsigned long long)(const void*)smem;

    v8f acc[2][4];
    #pragma unroll
    for (int i = 0; i < 2; ++i)
        #pragma unroll
        for (int j = 0; j < 4; ++j)
            #pragma unroll
            for (int r = 0; r < 8; ++r) acc[i][j][r] = 0.f;

    // ---- staging assignments (16B per lane per copy). Out-of-range pattern
    // rows are CLAMPED to PN-1; those columns masked to -inf in the epilogue.
    const int aRow = t >> 2;              // 0..63
    const int aOff = (t & 3) * 8;         // ushort offset within 32-wide slab
    const unsigned short* aSrc = qbf + (size_t)(qBase + aRow) * KPAD + aOff;
    const unsigned aDst = smemAddr + (unsigned)((aRow * 32 + aOff) * 2);

    unsigned bDst[4];
    const unsigned short* bSrc[4];
    #pragma unroll
    for (int i = 0; i < 4; ++i) {
        const int slot = t + i * 256;     // 0..1023 -> 256 rows x 4 slots
        const int bRow = slot >> 2;
        const int bOff = (slot & 3) * 8;
        const int gRow = min(nBase + bRow, PN - 1);
        bSrc[i] = dbf + (size_t)gRow * KPAD + bOff;
        bDst[i] = smemAddr + (unsigned)((bRow * 32 + bOff) * 2);
    }

    // ---- per-lane fragment addressing (ISA 16-bit A 16x32 / B 32x16 layouts)
    const int lm = lane & 15;
    const int kh = (lane >> 4) * 8;       // A: K half-select (ushorts)
    const int kb = (lane >> 4) * 8;       // B: K half-select (ushorts)

    // ---- prologue: async-stage slab 0 into buffer 0
    async_b128(aDst + LDS_A0, aSrc);
    #pragma unroll
    for (int i = 0; i < 4; ++i) async_b128(bDst[i] + LDS_B0, bSrc[i]);

    for (int kt = 0; kt < KSTEPS; ++kt) {
        wait_async0();     // own copies into buf[kt&1] (issued last iter) done
        __syncthreads();   // all waves' copies visible; prior reads retired

        // issue async DMA for slab kt+1 into the other buffer; the readers of
        // that buffer (iteration kt-1) all retired before this barrier.
        if (kt + 1 < KSTEPS) {
            const unsigned aBuf = (kt & 1) ? LDS_A0 : LDS_A1;
            const unsigned bBuf = (kt & 1) ? LDS_B0 : LDS_B1;
            const size_t kOfs = (size_t)(kt + 1) * 32;
            async_b128(aDst + aBuf, aSrc + kOfs);
            #pragma unroll
            for (int i = 0; i < 4; ++i) async_b128(bDst[i] + bBuf, bSrc[i] + kOfs);
        }
        __builtin_amdgcn_sched_barrier(0);   // DMA issue stays above the WMMAs

        // compute slab kt from buf[kt&1]
        const unsigned char* cA = smem + ((kt & 1) ? LDS_A1 : LDS_A0);
        const unsigned char* cB = smem + ((kt & 1) ? LDS_B1 : LDS_B0);

        Frag a[2];
        #pragma unroll
        for (int i = 0; i < 2; ++i) {
            const int arow = (mq * 2 + i) * 16 + lm;
            a[i].u[0] = *(const uint4*)(cA + (arow * 32 + kh) * 2);       // K kh..kh+7
            a[i].u[1] = *(const uint4*)(cA + (arow * 32 + kh + 16) * 2);  // K kh+16..+23
        }
        Frag b[4];
        #pragma unroll
        for (int j = 0; j < 4; ++j) {
            const int bcol = ng * 64 + j * 16 + lm;
            b[j].u[0] = *(const uint4*)(cB + (bcol * 32 + kb) * 2);       // K kb..kb+7
            b[j].u[1] = *(const uint4*)(cB + (bcol * 32 + kb + 16) * 2);  // K kb+16..+23
        }
        #pragma unroll
        for (int i = 0; i < 2; ++i)
            #pragma unroll
            for (int j = 0; j < 4; ++j)
                acc[i][j] = __builtin_amdgcn_wmma_f32_16x16x32_bf16(
                    false, a[i].v, false, b[j].v, (short)0, acc[i][j], false, false);
    }

    // ---- spill accumulators to LDS (overwrites staging, barrier-guarded)
    __syncthreads();
    #pragma unroll
    for (int i = 0; i < 2; ++i) {
        #pragma unroll
        for (int j = 0; j < 4; ++j) {
            const int col = ng * 64 + j * 16 + lm;
            #pragma unroll
            for (int r = 0; r < 8; ++r) {
                const int rowm = (mq * 2 + i) * 16 + r + ((lane >> 4) * 8);
                simLds[rowm * BN + col] = acc[i][j][r];
            }
        }
    }
    __syncthreads();

    // ---- fused per-tile top-10: one thread per query row
    if (t < BM) {
        float tv[TOPK]; int ti[TOPK];
        #pragma unroll
        for (int j = 0; j < TOPK; ++j) { tv[j] = NEG_INF; ti[j] = -1; }
        const float* row = simLds + t * BN;
        for (int c = 0; c < BN; ++c) {
            const int gid = nBase + c;
            const float v = (gid < PN) ? row[c] : NEG_INF;
            if (v > tv[TOPK - 1]) {
                int pos = TOPK - 1;
                while (pos > 0 && v > tv[pos - 1]) {
                    tv[pos] = tv[pos - 1]; ti[pos] = ti[pos - 1]; --pos;
                }
                tv[pos] = v; ti[pos] = gid;
            }
        }
        const size_t base = ((size_t)(qBase + t) * NCHUNK + blockIdx.x) * TOPK;
        #pragma unroll
        for (int j = 0; j < TOPK; ++j) { candV[base + j] = tv[j]; candI[base + j] = ti[j]; }
    }
}

// -------------------- final candidate merge --------------------------------
__global__ void final_topk(const float* __restrict__ candV,
                           const int* __restrict__ candI,
                           float* __restrict__ out) {
    const int q = blockIdx.x * blockDim.x + threadIdx.x;
    if (q >= QN) return;
    float tv[TOPK]; int ti[TOPK];
    #pragma unroll
    for (int j = 0; j < TOPK; ++j) { tv[j] = NEG_INF; ti[j] = -1; }
    const float* cv = candV + (size_t)q * NCHUNK * TOPK;
    const int*   ci = candI + (size_t)q * NCHUNK * TOPK;
    for (int c = 0; c < NCHUNK * TOPK; ++c) {
        const float v = cv[c];
        if (v > tv[TOPK - 1]) {
            const int id = ci[c];
            int pos = TOPK - 1;
            while (pos > 0 && v > tv[pos - 1]) {
                tv[pos] = tv[pos - 1]; ti[pos] = ti[pos - 1]; --pos;
            }
            tv[pos] = v; ti[pos] = id;
        }
    }
    #pragma unroll
    for (int j = 0; j < TOPK; ++j) out[(size_t)q * TOPK + j] = (float)ti[j];
}

// ---------------------------------------------------------------------------
extern "C" void kernel_launch(void* const* d_in, const int* in_sizes, int n_in,
                              void* d_out, int out_size, void* d_ws, size_t ws_size,
                              hipStream_t stream) {
    const float* patterns = (const float*)d_in[0];   // [30000, 3600]
    const float* expdata  = (const float*)d_in[1];   // [2048, 3600]
    // d_in[2] is topk==10 (compile-time constant here)

    unsigned char* ws = (unsigned char*)d_ws;
    const size_t QBF_BYTES = (size_t)QN * KPAD * 2;                 // ~14.8 MB
    const size_t DBF_BYTES = (size_t)PN * KPAD * 2;                 // ~217 MB
    const size_t CV_BYTES  = (size_t)QN * NCHUNK * TOPK * 4;        // ~9.7 MB

    unsigned short* qbf   = (unsigned short*)(ws);
    unsigned short* dbf   = (unsigned short*)(ws + QBF_BYTES);
    float*          candV = (float*)(ws + QBF_BYTES + DBF_BYTES);
    int*            candI = (int*)(ws + QBF_BYTES + DBF_BYTES + CV_BYTES);

    prep_q<<<QN, 256, 0, stream>>>(expdata, qbf);
    prep_d<<<PN, 256, 0, stream>>>(patterns, dbf);

    dim3 grid(NCHUNK, QN / BM);   // (118, 32)
    gemm_topk<<<grid, 256, 0, stream>>>(qbf, dbf, candV, candI);

    final_topk<<<(QN + 255) / 256, 256, 0, stream>>>(candV, candI, (float*)d_out);
}
